// GNNModel_5394478924266
// MI455X (gfx1250) — matrix operand order repaired and checked by reading, not compiled, not run on hardware
//
#include <hip/hip_runtime.h>
#include <hip/hip_bf16.h>

typedef __attribute__((ext_vector_type(16))) _Float16 v16h;
typedef __attribute__((ext_vector_type(8)))  float    v8f;

#define D 64
#define WARP 32
#define WROW 72   // padded LDS row stride in halves (36 dwords -> conflict-free)

// ---------------------------------------------------------------------------
// Degree / normalization
// ---------------------------------------------------------------------------
__global__ void k_deg_init(float* __restrict__ deg, int n) {
    int i = blockIdx.x * blockDim.x + threadIdx.x;
    if (i < n) deg[i] = 1.0f;                 // self-loop contributes 1
}

__global__ void k_deg_edges(const int* __restrict__ dst, float* __restrict__ deg, int e) {
    int i = blockIdx.x * blockDim.x + threadIdx.x;
    if (i < e) atomicAdd(&deg[dst[i]], 1.0f);
}

__global__ void k_rsqrt(float* __restrict__ deg, int n) {
    int i = blockIdx.x * blockDim.x + threadIdx.x;
    if (i < n) deg[i] = rsqrtf(deg[i]);       // deg >= 1 always
}

// ---------------------------------------------------------------------------
// GEMM: C[N x 64] = (RELU? max(A,0):A)[N x 64] @ W[64 x 64]  via WMMA f16->f32
// W is staged once per block into LDS, f16 + transposed + padded, so each
// B fragment is one aligned 32-byte LDS read. A rows load as float4.
// One wave computes a 16-row strip: 4 tiles of 16 cols, K=64 as 2x K=32.
// ---------------------------------------------------------------------------
template <bool RELU>
__global__ void k_gemm64(const float* __restrict__ A, const float* __restrict__ W,
                         float* __restrict__ C, int nStrips) {
    __shared__ _Float16 Wt[D * WROW];        // 4608 halves = 9 KB

    // cooperative stage: Wt[n][k] = (f16) W[k][n]
    for (int t = threadIdx.x; t < D * D; t += blockDim.x) {
        int k = t >> 6, n = t & 63;
        Wt[n * WROW + k] = (_Float16)W[t];
    }
    __syncthreads();

    const int lane  = threadIdx.x & (WARP - 1);
    const int wib   = threadIdx.x >> 5;                       // wave in block
    const int strip = blockIdx.x * (blockDim.x >> 5) + wib;   // 16-row strip
    if (strip >= nStrips) return;                             // wave-uniform exit

    const int m0  = strip * 16;
    const int rlo = lane & 15;      // row within strip for A; col within tile for B/C
    const int hi  = lane >> 4;      // lane group 0/1

    // ---- A fragments (16x32 f16 layout), two K-steps (kk = 0, 32) ----
    const float4* Arow4 = (const float4*)(A + (size_t)(m0 + rlo) * D);
    v16h a[2];
#pragma unroll
    for (int g = 0; g < 4; ++g) {            // K groups 32*(g>>1) + hi*16 + (g&1)*8
        float4 p = Arow4[(g >> 1) * 8 + hi * 4 + (g & 1) * 2];
        float4 q = Arow4[(g >> 1) * 8 + hi * 4 + (g & 1) * 2 + 1];
        if (RELU) {
            p.x = fmaxf(p.x, 0.f); p.y = fmaxf(p.y, 0.f);
            p.z = fmaxf(p.z, 0.f); p.w = fmaxf(p.w, 0.f);
            q.x = fmaxf(q.x, 0.f); q.y = fmaxf(q.y, 0.f);
            q.z = fmaxf(q.z, 0.f); q.w = fmaxf(q.w, 0.f);
        }
        const int ai = g >> 1;               // fragment (kk = 0 / 32)
        const int fi = (g & 1) * 8;          // element base (K or K+8 halves)
        a[ai][fi + 0] = (_Float16)p.x; a[ai][fi + 1] = (_Float16)p.y;
        a[ai][fi + 2] = (_Float16)p.z; a[ai][fi + 3] = (_Float16)p.w;
        a[ai][fi + 4] = (_Float16)q.x; a[ai][fi + 5] = (_Float16)q.y;
        a[ai][fi + 6] = (_Float16)q.z; a[ai][fi + 7] = (_Float16)q.w;
    }

    const int kg = hi * 16;         // B: lanes<16 hold K=0..15, lanes>=16 K=16..31
#pragma unroll
    for (int t = 0; t < 4; ++t) {   // 4 column tiles of 16
        const int n0 = t * 16;
        const _Float16* wt = &Wt[(n0 + rlo) * WROW];
        v16h b0 = *(const v16h*)(wt + kg);        // K = kg .. kg+15
        v16h b1 = *(const v16h*)(wt + 32 + kg);   // K = 32+kg .. 32+kg+15
        v8f c = {};
        c = __builtin_amdgcn_wmma_f32_16x16x32_f16(false, a[0], false, b0,
                                                   (short)0, c, false, false);
        c = __builtin_amdgcn_wmma_f32_16x16x32_f16(false, a[1], false, b1,
                                                   (short)0, c, false, false);
        // C layout: element r -> row m0 + r + hi*8, col n0 + rlo
        float* Crow = C + (size_t)(m0 + hi * 8) * D + n0 + rlo;
#pragma unroll
        for (int r = 0; r < 8; ++r) Crow[(size_t)r * D] = c[r];
    }
}

// ---------------------------------------------------------------------------
// Aggregation: out = bias + self-loop term (non-atomic), then edge scatter.
// ---------------------------------------------------------------------------
__global__ void k_agg_init(const float* __restrict__ h, const float* __restrict__ dinv,
                           const float* __restrict__ bias, float* __restrict__ out, int n64) {
    int idx = blockIdx.x * blockDim.x + threadIdx.x;
    if (idx < n64) {
        int i = idx >> 6, f = idx & 63;
        float di = dinv[i];
        out[idx] = bias[f] + h[idx] * di * di;
    }
}

__global__ void k_agg_edges(const int* __restrict__ src, const int* __restrict__ dst,
                            const float* __restrict__ h, const float* __restrict__ dinv,
                            float* __restrict__ out, int E) {
    int gid  = blockIdx.x * blockDim.x + threadIdx.x;
    int e    = gid >> 5;
    int lane = gid & 31;
    if (e >= E) return;
    int s = src[e], d = dst[e];                      // broadcast reads
    float norm = dinv[s] * dinv[d];
    const float2* hr2 = (const float2*)(h + (size_t)s * D);
    float2 v = hr2[lane];                            // feats 2*lane, 2*lane+1
    float* orow = out + (size_t)d * D + 2 * lane;
    atomicAdd(&orow[0], v.x * norm);
    atomicAdd(&orow[1], v.y * norm);
}

// ---------------------------------------------------------------------------
// Column-sum of relu(H) over all rows -> colsum[64]
// ---------------------------------------------------------------------------
__global__ void k_zero64(float* __restrict__ p) { p[threadIdx.x] = 0.0f; }

__global__ void k_colsum_relu(const float* __restrict__ h, float* __restrict__ colsum, int n) {
    __shared__ float sh[D];
    int f   = threadIdx.x & 63;
    int sub = threadIdx.x >> 6;                      // 4 sub-rows per block
    if (threadIdx.x < D) sh[threadIdx.x] = 0.0f;
    __syncthreads();
    float acc = 0.0f;
    for (int i = blockIdx.x * 4 + sub; i < n; i += gridDim.x * 4)
        acc += fmaxf(h[(size_t)i * D + f], 0.0f);
    atomicAdd(&sh[f], acc);
    __syncthreads();
    if (threadIdx.x < D) atomicAdd(&colsum[threadIdx.x], sh[threadIdx.x]);
}

// final: out[j] = bfc[j] + (colsum/N) . Wfc[:, j]
__global__ void k_final(const float* __restrict__ colsum, const float* __restrict__ Wfc,
                        const float* __restrict__ bfc, float* __restrict__ out, float invN) {
    int j = threadIdx.x;   // 64 threads
    float acc = 0.0f;
#pragma unroll
    for (int f = 0; f < D; ++f) acc += colsum[f] * Wfc[f * D + j];
    out[j] = bfc[j] + acc * invN;
}

// ---------------------------------------------------------------------------
extern "C" void kernel_launch(void* const* d_in, const int* in_sizes, int n_in,
                              void* d_out, int out_size, void* d_ws, size_t ws_size,
                              hipStream_t stream) {
    const float* x   = (const float*)d_in[0];
    const int*   ei  = (const int*)  d_in[1];
    const float* W1  = (const float*)d_in[2];
    const float* b1  = (const float*)d_in[3];
    const float* W2  = (const float*)d_in[4];
    const float* b2  = (const float*)d_in[5];
    const float* Wfc = (const float*)d_in[6];
    const float* bfc = (const float*)d_in[7];

    const int N = in_sizes[0] / D;          // 100000
    const int E = in_sizes[1] / 2;          // 1600000
    const int* src = ei;
    const int* dst = ei + E;

    float* ws     = (float*)d_ws;
    float* dinv   = ws;                     // N
    float* hbuf   = dinv + N;               // N*64  (16B-aligned: N*4 % 16 == 0)
    float* obuf   = hbuf + (size_t)N * D;   // N*64
    float* colsum = obuf + (size_t)N * D;   // 64

    const int nStrips  = N / 16;                        // 6250
    const int gemmBlks = (nStrips + 7) / 8;             // 8 waves/block
    const int n64      = N * D;

    // degrees / dinv
    k_deg_init <<<(N + 255) / 256, 256, 0, stream>>>(dinv, N);
    k_deg_edges<<<(E + 255) / 256, 256, 0, stream>>>(dst, dinv, E);
    k_rsqrt    <<<(N + 255) / 256, 256, 0, stream>>>(dinv, N);

    // layer 1: h = x @ W1 ; out1 = b1 + self + scatter ; (relu deferred)
    k_gemm64<false><<<gemmBlks, 256, 0, stream>>>(x, W1, hbuf, nStrips);
    k_agg_init <<<(n64 + 255) / 256, 256, 0, stream>>>(hbuf, dinv, b1, obuf, n64);
    k_agg_edges<<<((size_t)E * 32 + 255) / 256, 256, 0, stream>>>(src, dst, hbuf, dinv, obuf, E);

    // layer 2: h = relu(out1) @ W2 ; out2 = b2 + self + scatter
    k_gemm64<true><<<gemmBlks, 256, 0, stream>>>(obuf, W2, hbuf, nStrips);
    k_agg_init <<<(n64 + 255) / 256, 256, 0, stream>>>(hbuf, dinv, b2, obuf, n64);
    k_agg_edges<<<((size_t)E * 32 + 255) / 256, 256, 0, stream>>>(src, dst, hbuf, dinv, obuf, E);

    // FC + mean collapsed: mean_rows(relu(out2)) @ Wfc + bfc
    k_zero64<<<1, 64, 0, stream>>>(colsum);
    k_colsum_relu<<<1024, 256, 0, stream>>>(obuf, colsum, N);
    k_final<<<1, 64, 0, stream>>>(colsum, Wfc, bfc, (float*)d_out, 1.0f / (float)N);
}
